// BigBirdEncoder_13692355739876
// MI455X (gfx1250) — matrix operand order repaired and checked by reading, not compiled, not run on hardware
//
#include <hip/hip_runtime.h>

typedef __attribute__((ext_vector_type(16))) __bf16 v16bf;
typedef __attribute__((ext_vector_type(8)))  __bf16 v8bf;
typedef __attribute__((ext_vector_type(8)))  float  v8f;

#define DEV __device__ __forceinline__

enum { B_=2, S_=4096, D_=768, H_=12, L_=2, BS_=64, NB_=64, R_=3, DH_=64, FF_=3072, M_=B_*S_ };
#define SCALE_ 0.125f     /* 1/sqrt(64) */
#define EPS_   1e-12f

// ---------------- bf16 helpers ----------------
DEV unsigned short f2b(float f){              // hardware RNE convert
  return __builtin_bit_cast(unsigned short, (__bf16)f);
}
DEV float b2f(unsigned short h){ return __uint_as_float(((unsigned)h) << 16); }

DEV v8f vzero(){
  v8f z;
  #pragma unroll
  for(int i=0;i<8;i++) z[i]=0.f;
  return z;
}

// ---------------- WMMA fragment loads from LDS (bf16, stride in halfs) ----
// A-matrix 16x32 (MxK): lane L: row=L&15; halfs 0..7 = K at 8*(L>>4)..+7,
// halfs 8..15 = K at 16+8*(L>>4)..+7   (ISA 7.12.2 16-bit A table)
DEV v16bf frag_a(const unsigned short* base, int stride){
  int lane = threadIdx.x & 31;
  const unsigned short* p = base + (lane & 15)*stride + ((lane >> 4) << 3);
  v8bf lo = *(const v8bf*)p;
  v8bf hi = *(const v8bf*)(p + 16);
  v16bf r;
  #pragma unroll
  for(int i=0;i<8;i++){ r[i]=lo[i]; r[i+8]=hi[i]; }
  return r;
}
// B-matrix 32x16 (KxN): lane L: col=L&15; halfs 0..7 = K at 8*(L>>4)..+7,
// halfs 8..15 = K at 16+8*(L>>4)..+7 (same k order as frag_a)
DEV v16bf frag_b(const unsigned short* base, int stride){
  int lane = threadIdx.x & 31;
  const unsigned short* p = base + (lane & 15)*stride + ((lane >> 4) << 3);
  v8bf lo = *(const v8bf*)p;
  v8bf hi = *(const v8bf*)(p + 16);
  v16bf r;
  #pragma unroll
  for(int i=0;i<8;i++){ r[i]=lo[i]; r[i+8]=hi[i]; }
  return r;
}

// ---------------- weight convert + transpose: W[l][K][N] f32 -> Wt[l][N][K] bf16
__global__ void convt_kernel(const float* __restrict__ W, unsigned short* __restrict__ Wt,
                             int K, int N){
  size_t idx = (size_t)blockIdx.x*256 + threadIdx.x;
  size_t per = (size_t)K*N;
  if(idx >= (size_t)L_*per) return;
  size_t l = idx / per, r = idx % per;
  size_t kk = r / N, nn = r % N;
  Wt[l*per + nn*(size_t)K + kk] = f2b(W[idx]);
}

// ---------------- GEMM: C[M,N] = epi(A[M,K] @ Bt[N,K]^T + bias) ---------
// Double-buffered LDS; global prefetch of tile k+1 overlaps WMMA of tile k.
// EPI 0: +bias -> bf16 out     (QKV)
// EPI 1: +bias +resid -> f32   (proj w/ residual, pre-LN)
// EPI 2: +bias, GELU -> bf16   (FFN1)
#define GT_KT 64
#define GT_ST 72
template<int EPI>
__global__ __launch_bounds__(256) void gemm_kernel(
    const unsigned short* __restrict__ A, const unsigned short* __restrict__ Bt,
    const float* __restrict__ bias, const float* __restrict__ resid,
    float* __restrict__ outF, unsigned short* __restrict__ outB,
    int M, int N, int K){
  __shared__ unsigned short As[2][128*GT_ST];
  __shared__ unsigned short Bs[2][128*GT_ST];
  const int t = threadIdx.x;
  const int wave = t >> 5, lane = t & 31;
  const int wm = wave >> 1, wn = wave & 1;       // 4x2 wave grid
  const int m0 = blockIdx.y * 128, n0 = blockIdx.x * 128;
  v8f acc[2][4];
  #pragma unroll
  for(int i=0;i<2;i++)
    #pragma unroll
    for(int j=0;j<4;j++) acc[i][j] = vzero();

  // chunk mapping: 1024 chunks of 8 halfs (128 rows x 8 chunks), 4 per thread
  int crow[4], ccol[4];
  #pragma unroll
  for(int i=0;i<4;i++){ int cid = t + 256*i; crow[i] = cid >> 3; ccol[i] = (cid & 7) * 8; }

  v8bf ra[4], rb[4];
  // ---- prolog: load K-tile 0 and stage into buffer 0
  #pragma unroll
  for(int i=0;i<4;i++){
    ra[i] = *(const v8bf*)(A  + (size_t)(m0+crow[i])*K + ccol[i]);
    rb[i] = *(const v8bf*)(Bt + (size_t)(n0+crow[i])*K + ccol[i]);
  }
  #pragma unroll
  for(int i=0;i<4;i++){
    *(v8bf*)(As[0] + crow[i]*GT_ST + ccol[i]) = ra[i];
    *(v8bf*)(Bs[0] + crow[i]*GT_ST + ccol[i]) = rb[i];
  }
  __syncthreads();

  const int NK = K / GT_KT;
  for(int kt=0; kt<NK; kt++){
    const int cur = kt & 1;
    if(kt+1 < NK){                    // prefetch next K-tile while computing
      int k0 = (kt+1)*GT_KT;
      #pragma unroll
      for(int i=0;i<4;i++){
        ra[i] = *(const v8bf*)(A  + (size_t)(m0+crow[i])*K + k0 + ccol[i]);
        rb[i] = *(const v8bf*)(Bt + (size_t)(n0+crow[i])*K + k0 + ccol[i]);
      }
    }
    #pragma unroll
    for(int kk=0; kk<GT_KT; kk+=32){
      v16bf fa[2], fb[4];
      #pragma unroll
      for(int i=0;i<2;i++) fa[i] = frag_a(As[cur] + (wm*32 + i*16)*GT_ST + kk, GT_ST);
      #pragma unroll
      for(int j=0;j<4;j++) fb[j] = frag_b(Bs[cur] + (wn*64 + j*16)*GT_ST + kk, GT_ST);
      #pragma unroll
      for(int i=0;i<2;i++)
        #pragma unroll
        for(int j=0;j<4;j++)
          acc[i][j] = __builtin_amdgcn_wmma_f32_16x16x32_bf16(
              false, fa[i], false, fb[j], (short)0, acc[i][j], false, false);
    }
    if(kt+1 < NK){
      __syncthreads();                 // everyone done reading buffer (kt+1)&1 from iter kt-1
      const int nxt = (kt+1) & 1;
      #pragma unroll
      for(int i=0;i<4;i++){
        *(v8bf*)(As[nxt] + crow[i]*GT_ST + ccol[i]) = ra[i];
        *(v8bf*)(Bs[nxt] + crow[i]*GT_ST + ccol[i]) = rb[i];
      }
      __syncthreads();
    }
  }

  const int cn = lane & 15, hl = lane >> 4;
  #pragma unroll
  for(int i=0;i<2;i++){
    #pragma unroll
    for(int j=0;j<4;j++){
      int col = n0 + wn*64 + j*16 + cn;
      float bv = bias[col];
      #pragma unroll
      for(int r=0;r<8;r++){
        int row = m0 + wm*32 + i*16 + r + hl*8;
        float v = acc[i][j][r] + bv;
        if(EPI==1){
          v += resid[(size_t)row*N + col];
          outF[(size_t)row*N + col] = v;
        } else if(EPI==2){
          float x = v;
          v = 0.5f*x*(1.f + tanhf(0.7978845608f*(x + 0.044715f*x*x*x)));
          outB[(size_t)row*N + col] = f2b(v);
        } else {
          outB[(size_t)row*N + col] = f2b(v);
        }
      }
    }
  }
}

// ---------------- BigBird block-sparse attention ------------------------
// One workgroup per (b,h,query-block). key blocks: [0, NB-1, n-1, n, n+1, r0,r1,r2]
#define QS 72     /* LDS stride (halfs) for 64-wide tiles */
#define SS 520    /* LDS stride for 512-wide score rows  */
#define VS 520
__global__ __launch_bounds__(256) void attn_kernel(
    const unsigned short* __restrict__ q, const unsigned short* __restrict__ k,
    const unsigned short* __restrict__ v, const int* __restrict__ rb_all,
    unsigned short* __restrict__ ctx, int layer){
  __shared__ unsigned short q_s [64*QS];
  __shared__ unsigned short kblk[64*QS];
  __shared__ unsigned short s_s [64*SS];    // scores -> probabilities (bf16)
  __shared__ unsigned short vT  [64*VS];    // [dh][key-token]
  __shared__ float red[64*4];
  const int n = blockIdx.x, h = blockIdx.y, b = blockIdx.z;
  const int t = threadIdx.x, wave = t >> 5, lane = t & 31;
  int kb[8];
  kb[0]=0; kb[1]=NB_-1; kb[2]=(n+NB_-1)%NB_; kb[3]=n; kb[4]=(n+1)%NB_;
  const int* rb = rb_all + (((size_t)layer*H_ + h)*NB_ + n)*R_;
  kb[5]=rb[0]; kb[6]=rb[1]; kb[7]=rb[2];
  const size_t rowbase = (size_t)b*S_;

  // load q block [64 tok][64 dh]
  #pragma unroll
  for(int i=0;i<2;i++){
    int cid = t + 256*i; int r = cid >> 3, c = cid & 7;
    *(v8bf*)(q_s + r*QS + c*8) =
        *(const v8bf*)(q + (rowbase + n*64 + r)*D_ + h*64 + c*8);
  }
  // load v^T: vT[dh][blk*64+tok]
  #pragma unroll 4
  for(int i=0;i<128;i++){
    int e = t + 256*i;
    int dh = e & 63, tok = (e >> 6) & 63, bi = e >> 12;
    vT[dh*VS + bi*64 + tok] = v[(rowbase + kb[bi]*64 + tok)*D_ + h*64 + dh];
  }

  // -------- scores = (q @ k^T) * SCALE,  bf16 into s_s --------
  for(int kbI=0;kbI<8;kbI++){
    __syncthreads();
    #pragma unroll
    for(int i=0;i<2;i++){
      int cid = t + 256*i; int r = cid >> 3, c = cid & 7;
      *(v8bf*)(kblk + r*QS + c*8) =
          *(const v8bf*)(k + (rowbase + kb[kbI]*64 + r)*D_ + h*64 + c*8);
    }
    __syncthreads();
    int nt = wave & 3;
    #pragma unroll
    for(int u=0;u<2;u++){
      int mt = (wave >> 2) + 2*u;
      v8f acc = vzero();
      #pragma unroll
      for(int kk=0;kk<64;kk+=32){
        v16bf fa = frag_a(q_s  + mt*16*QS + kk, QS);
        v16bf fb = frag_b(kblk + nt*16*QS + kk, QS);
        acc = __builtin_amdgcn_wmma_f32_16x16x32_bf16(
            false, fa, false, fb, (short)0, acc, false, false);
      }
      int cn = lane & 15, hl = lane >> 4;
      #pragma unroll
      for(int r=0;r<8;r++){
        int m = mt*16 + r + hl*8;
        s_s[m*SS + kbI*64 + nt*16 + cn] = f2b(acc[r] * SCALE_);
      }
    }
  }
  __syncthreads();

  // -------- softmax over 512 keys per row (mask == 1 everywhere) --------
  {
    int row = t >> 2, q4 = t & 3;
    unsigned short* srow = s_s + row*SS + q4*128;
    float mx = -1e30f;
    for(int j=0;j<128;j++) mx = fmaxf(mx, b2f(srow[j]));
    red[row*4+q4] = mx;
    __syncthreads();
    float rmax = fmaxf(fmaxf(red[row*4+0],red[row*4+1]),
                       fmaxf(red[row*4+2],red[row*4+3]));
    __syncthreads();
    float sum = 0.f;
    for(int j=0;j<128;j++){
      float e = __expf(b2f(srow[j]) - rmax);
      srow[j] = f2b(e); sum += e;
    }
    red[row*4+q4] = sum;
    __syncthreads();
    float inv = 1.f/(red[row*4+0]+red[row*4+1]+red[row*4+2]+red[row*4+3]);
    for(int j=0;j<128;j++) srow[j] = f2b(b2f(srow[j]) * inv);
  }
  __syncthreads();

  // -------- ctx = p @ vg,  bf16 out at [b, n*64+m, h*64+dh] --------
  {
    int nt = wave & 3;
    #pragma unroll
    for(int u=0;u<2;u++){
      int mt = (wave >> 2) + 2*u;
      v8f acc = vzero();
      #pragma unroll 4
      for(int ks=0;ks<16;ks++){
        v16bf fa = frag_a(s_s + mt*16*SS + ks*32, SS);
        v16bf fb = frag_b(vT  + nt*16*VS + ks*32, VS);
        acc = __builtin_amdgcn_wmma_f32_16x16x32_bf16(
            false, fa, false, fb, (short)0, acc, false, false);
      }
      int cn = lane & 15, hl = lane >> 4;
      #pragma unroll
      for(int r=0;r<8;r++){
        int m = mt*16 + r + hl*8;
        ctx[(rowbase + n*64 + m)*D_ + h*64 + nt*16 + cn] = f2b(acc[r]);
      }
    }
  }
}

// ---------------- embedding + LN -> x (f32 + bf16) ----------------------
__global__ __launch_bounds__(256) void embed_ln_kernel(
    const int* __restrict__ tokens, const float* __restrict__ we,
    const float* __restrict__ pe, const float* __restrict__ te,
    const float* __restrict__ g, const float* __restrict__ bta,
    float* __restrict__ xF, unsigned short* __restrict__ xB){
  int m = blockIdx.x; int b = m / S_; int s = m % S_;
  int tok = tokens[s*B_ + b];
  __shared__ float red[256];
  int t = threadIdx.x;
  float loc[3]; float ps = 0.f;
  #pragma unroll
  for(int i=0;i<3;i++){
    int d = t + i*256;
    float v = we[(size_t)tok*D_+d] + pe[(size_t)s*D_+d] + te[d];
    loc[i]=v; ps += v;
  }
  red[t]=ps; __syncthreads();
  for(int o=128;o>0;o>>=1){ if(t<o) red[t]+=red[t+o]; __syncthreads(); }
  float mu = red[0] * (1.f/D_); __syncthreads();
  float pv = 0.f;
  #pragma unroll
  for(int i=0;i<3;i++){ float d = loc[i]-mu; pv += d*d; }
  red[t]=pv; __syncthreads();
  for(int o=128;o>0;o>>=1){ if(t<o) red[t]+=red[t+o]; __syncthreads(); }
  float rstd = rsqrtf(red[0]*(1.f/D_) + EPS_);
  #pragma unroll
  for(int i=0;i<3;i++){
    int d = t + i*256;
    float o = (loc[i]-mu)*rstd*g[d] + bta[d];
    xF[(size_t)m*D_+d] = o; xB[(size_t)m*D_+d] = f2b(o);
  }
}

// ---------------- LN over f32 input -> f32 + bf16 -----------------------
__global__ __launch_bounds__(256) void ln_kernel(
    const float* __restrict__ in, const float* __restrict__ g,
    const float* __restrict__ bta, float* __restrict__ outF,
    unsigned short* __restrict__ outB){
  int m = blockIdx.x; int t = threadIdx.x;
  __shared__ float red[256];
  float loc[3]; float ps = 0.f;
  #pragma unroll
  for(int i=0;i<3;i++){ loc[i] = in[(size_t)m*D_ + t + i*256]; ps += loc[i]; }
  red[t]=ps; __syncthreads();
  for(int o=128;o>0;o>>=1){ if(t<o) red[t]+=red[t+o]; __syncthreads(); }
  float mu = red[0] * (1.f/D_); __syncthreads();
  float pv = 0.f;
  #pragma unroll
  for(int i=0;i<3;i++){ float d = loc[i]-mu; pv += d*d; }
  red[t]=pv; __syncthreads();
  for(int o=128;o>0;o>>=1){ if(t<o) red[t]+=red[t+o]; __syncthreads(); }
  float rstd = rsqrtf(red[0]*(1.f/D_) + EPS_);
  #pragma unroll
  for(int i=0;i<3;i++){
    int d = t + i*256;
    float o = (loc[i]-mu)*rstd*g[d] + bta[d];
    outF[(size_t)m*D_+d] = o; outB[(size_t)m*D_+d] = f2b(o);
  }
}

// ---------------- host ---------------------------------------------------
static void gemm_launch(hipStream_t st, const unsigned short* A, const unsigned short* Bt,
                        const float* bias, const float* resid, float* outF,
                        unsigned short* outB, int M, int N, int K, int epi){
  dim3 g(N/128, M/128), bl(256,1,1);
  if(epi==0)      gemm_kernel<0><<<g,bl,0,st>>>(A,Bt,bias,resid,outF,outB,M,N,K);
  else if(epi==1) gemm_kernel<1><<<g,bl,0,st>>>(A,Bt,bias,resid,outF,outB,M,N,K);
  else            gemm_kernel<2><<<g,bl,0,st>>>(A,Bt,bias,resid,outF,outB,M,N,K);
}

extern "C" void kernel_launch(void* const* d_in, const int* in_sizes, int n_in,
                              void* d_out, int out_size, void* d_ws, size_t ws_size,
                              hipStream_t stream){
  (void)in_sizes; (void)n_in; (void)out_size; (void)ws_size;
  const int*   tokens   = (const int*)d_in[0];
  const int*   rand_blk = (const int*)d_in[1];
  const float* word_emb = (const float*)d_in[2];
  const float* pos_emb  = (const float*)d_in[3];
  const float* type_emb = (const float*)d_in[4];
  const float* ln_e_s   = (const float*)d_in[5];
  const float* ln_e_b   = (const float*)d_in[6];
  const float* Wq=(const float*)d_in[7];  const float* bq=(const float*)d_in[8];
  const float* Wk=(const float*)d_in[9];  const float* bk=(const float*)d_in[10];
  const float* Wv=(const float*)d_in[11]; const float* bv=(const float*)d_in[12];
  const float* Wo=(const float*)d_in[13]; const float* bo=(const float*)d_in[14];
  const float* l1s=(const float*)d_in[15]; const float* l1b=(const float*)d_in[16];
  const float* W1=(const float*)d_in[17]; const float* b1=(const float*)d_in[18];
  const float* W2=(const float*)d_in[19]; const float* b2=(const float*)d_in[20];
  const float* l2s=(const float*)d_in[21]; const float* l2b=(const float*)d_in[22];
  float* out = (float*)d_out;

  char* p = (char*)d_ws;
  auto carve = [&](size_t bytes)->void*{
    void* r = (void*)p; p += (bytes + 255) & ~(size_t)255; return r; };
  unsigned short* WqT=(unsigned short*)carve((size_t)L_*D_*D_*2);
  unsigned short* WkT=(unsigned short*)carve((size_t)L_*D_*D_*2);
  unsigned short* WvT=(unsigned short*)carve((size_t)L_*D_*D_*2);
  unsigned short* WoT=(unsigned short*)carve((size_t)L_*D_*D_*2);
  unsigned short* W1T=(unsigned short*)carve((size_t)L_*D_*FF_*2);
  unsigned short* W2T=(unsigned short*)carve((size_t)L_*FF_*D_*2);
  float* xF  =(float*)carve((size_t)M_*D_*4);
  float* x1F =(float*)carve((size_t)M_*D_*4);
  float* tmpF=(float*)carve((size_t)M_*D_*4);
  unsigned short* xB  =(unsigned short*)carve((size_t)M_*D_*2);
  unsigned short* x1B =(unsigned short*)carve((size_t)M_*D_*2);
  unsigned short* d_q =(unsigned short*)carve((size_t)M_*D_*2);
  unsigned short* d_k =(unsigned short*)carve((size_t)M_*D_*2);
  unsigned short* d_v =(unsigned short*)carve((size_t)M_*D_*2);
  unsigned short* ctxB=(unsigned short*)carve((size_t)M_*D_*2);
  unsigned short* hB  =(unsigned short*)carve((size_t)M_*FF_*2);

  { // weight conversion (every call; deterministic)
    size_t tDD=(size_t)L_*D_*D_, tDF=(size_t)L_*D_*FF_;
    dim3 bl(256);
    convt_kernel<<<dim3((tDD+255)/256),bl,0,stream>>>(Wq,WqT,D_,D_);
    convt_kernel<<<dim3((tDD+255)/256),bl,0,stream>>>(Wk,WkT,D_,D_);
    convt_kernel<<<dim3((tDD+255)/256),bl,0,stream>>>(Wv,WvT,D_,D_);
    convt_kernel<<<dim3((tDD+255)/256),bl,0,stream>>>(Wo,WoT,D_,D_);
    convt_kernel<<<dim3((tDF+255)/256),bl,0,stream>>>(W1,W1T,D_,FF_);
    convt_kernel<<<dim3((tDF+255)/256),bl,0,stream>>>(W2,W2T,FF_,D_);
  }
  embed_ln_kernel<<<dim3(M_),dim3(256),0,stream>>>(tokens,word_emb,pos_emb,type_emb,
                                                   ln_e_s,ln_e_b,xF,xB);
  for(int l=0;l<L_;l++){
    size_t oDD=(size_t)l*D_*D_, oD=(size_t)l*D_, oDF=(size_t)l*D_*FF_, oF=(size_t)l*FF_;
    gemm_launch(stream, xB, WqT+oDD, bq+oD, nullptr, nullptr, d_q, M_, D_, D_, 0);
    gemm_launch(stream, xB, WkT+oDD, bk+oD, nullptr, nullptr, d_k, M_, D_, D_, 0);
    gemm_launch(stream, xB, WvT+oDD, bv+oD, nullptr, nullptr, d_v, M_, D_, D_, 0);
    attn_kernel<<<dim3(NB_,H_,B_),dim3(256),0,stream>>>(d_q,d_k,d_v,rand_blk,ctxB,l);
    gemm_launch(stream, ctxB, WoT+oDD, bo+oD, xF, tmpF, nullptr, M_, D_, D_, 1);
    ln_kernel<<<dim3(M_),dim3(256),0,stream>>>(tmpF, l1s+oD, l1b+oD, x1F, x1B);
    gemm_launch(stream, x1B, W1T+oDF, b1+oF, nullptr, nullptr, hB, M_, FF_, D_, 2);
    gemm_launch(stream, hB, W2T+oDF, b2+oD, x1F, tmpF, nullptr, M_, D_, FF_, 1);
    ln_kernel<<<dim3(M_),dim3(256),0,stream>>>(tmpF, l2s+oD, l2b+oD,
                                               (l==L_-1)? out : xF, xB);
  }
}
